// DeltaRuleMixer_67843303407960
// MI455X (gfx1250) — matrix operand order repaired and checked by reading, not compiled, not run on hardware
//
#include <hip/hip_runtime.h>
#include <hip/hip_bf16.h>

// ---------------------------------------------------------------------------
// DeltaRuleMixer for MI455X (gfx1250, wave32, WMMA).
//
// Roofline: 5x [8192x1024x1024] GEMMs = ~86 GFLOP over ~0.5 GB of traffic.
// At 23.3 TB/s the memory floor is ~21us; fp32 VALU would make compute the
// bottleneck, so the GEMMs are run as bf16 WMMA (v_wmma_f32_16x16x32_bf16,
// fp32 accumulate). The sequential delta-rule scan (3.2 GFLOP, rank-1 state
// updates) runs as fp32 VALU with LDS-staged chunks, one workgroup per (b,h).
// Activations (sigmoid/silu) use raw v_rcp_f32 instead of IEEE div chains.
// ---------------------------------------------------------------------------

#define D_MODEL 1024
#define N_HEADS 16
#define HEAD_DIM 64
#define BATCH 4
#define SEQ 2048
#define ROWS (BATCH * SEQ)   // 8192

typedef __attribute__((ext_vector_type(16))) __bf16 bf16x16;
typedef __attribute__((ext_vector_type(8)))  __bf16 bf16x8;
typedef __attribute__((ext_vector_type(8)))  float  f32x8;

__device__ __forceinline__ unsigned short f2bf(float f) {
  unsigned int u = __float_as_uint(f);
  unsigned int lsb = (u >> 16) & 1u;
  u += 0x7fffu + lsb;              // round-to-nearest-even
  return (unsigned short)(u >> 16);
}

__device__ __forceinline__ float fast_sigmoid(float x) {
  // v_exp_f32 + v_add + v_rcp_f32 (no IEEE division chain)
  return __builtin_amdgcn_rcpf(1.f + __expf(-x));
}

// ---------------------------------------------------------------------------
// fp32 -> bf16 elementwise convert
// ---------------------------------------------------------------------------
__global__ __launch_bounds__(256) void cvt_f32_bf16_kernel(const float* __restrict__ in,
                                                           unsigned short* __restrict__ out,
                                                           int n) {
  int i = blockIdx.x * 256 + threadIdx.x;
  if (i < n) out[i] = f2bf(in[i]);
}

// ---------------------------------------------------------------------------
// W[K][N] fp32 -> Wt[N][K] bf16 (LDS-tiled transpose, 32x32 tiles)
// ---------------------------------------------------------------------------
__global__ __launch_bounds__(256) void transpose_cvt_kernel(const float* __restrict__ W,
                                                            unsigned short* __restrict__ Wt,
                                                            int K, int N) {
  __shared__ unsigned short tile[32][33];
  int tx = threadIdx.x & 31, ty = threadIdx.x >> 5;   // 32 x 8
  int bn = blockIdx.x * 32, bk = blockIdx.y * 32;
  #pragma unroll
  for (int i = 0; i < 32; i += 8) {
    int k = bk + ty + i, n = bn + tx;
    tile[ty + i][tx] = f2bf(W[(size_t)k * N + n]);
  }
  __syncthreads();
  #pragma unroll
  for (int i = 0; i < 32; i += 8) {
    int n = bn + ty + i, k = bk + tx;
    Wt[(size_t)n * K + k] = tile[tx][ty + i];
  }
}

// ---------------------------------------------------------------------------
// C[M][N] = A[M][K](bf16, row-major) @ Bt[N][K](bf16, K-major)   fp32 out
// Wave tile 64x64 (16 WMMA accum tiles), 4 waves -> 128x128 per workgroup.
// epi: 0 = none, 1 = silu
// ---------------------------------------------------------------------------
__device__ __forceinline__ bf16x16 load_a_frag(const __bf16* __restrict__ A,
                                               int row, int K, int kk, int half) {
  const __bf16* p = A + (size_t)row * K + kk + half * 8;
  bf16x8 lo = *(const bf16x8*)p;          // K = kk + half*8 .. +7
  bf16x8 hi = *(const bf16x8*)(p + 16);   // K = kk + 16 + half*8 .. +7
  return __builtin_shufflevector(lo, hi, 0, 1, 2, 3, 4, 5, 6, 7,
                                         8, 9, 10, 11, 12, 13, 14, 15);
}

__global__ __launch_bounds__(128) void gemm_bf16_kernel(const unsigned short* __restrict__ A_,
                                                        const unsigned short* __restrict__ Bt_,
                                                        float* __restrict__ C,
                                                        int M, int N, int K, int epi) {
  const __bf16* A  = (const __bf16*)A_;
  const __bf16* Bt = (const __bf16*)Bt_;
  int lane = threadIdx.x & 31;
  int wave = threadIdx.x >> 5;
  int half = lane >> 4;
  int lrow = lane & 15;
  int wm = wave & 1, wn = wave >> 1;
  int m0 = blockIdx.x * 128 + wm * 64;
  int n0 = blockIdx.y * 128 + wn * 64;

  f32x8 acc[4][4];
  #pragma unroll
  for (int mi = 0; mi < 4; ++mi)
    #pragma unroll
    for (int ni = 0; ni < 4; ++ni)
      #pragma unroll
      for (int r = 0; r < 8; ++r) acc[mi][ni][r] = 0.f;

  for (int kk = 0; kk < K; kk += 32) {
    bf16x16 a[4], b[4];
    #pragma unroll
    for (int mi = 0; mi < 4; ++mi)
      a[mi] = load_a_frag(A, m0 + mi * 16 + lrow, K, kk, half);
    #pragma unroll
    for (int ni = 0; ni < 4; ++ni)
      b[ni] = load_a_frag(Bt, n0 + ni * 16 + lrow, K, kk, half);
    #pragma unroll
    for (int mi = 0; mi < 4; ++mi)
      #pragma unroll
      for (int ni = 0; ni < 4; ++ni)
        acc[mi][ni] = __builtin_amdgcn_wmma_f32_16x16x32_bf16(
            /*neg_a=*/false, a[mi], /*neg_b=*/false, b[ni],
            /*c_mod=*/(short)0, acc[mi][ni], /*reuse_a=*/false, /*reuse_b=*/false);
  }

  #pragma unroll
  for (int mi = 0; mi < 4; ++mi)
    #pragma unroll
    for (int ni = 0; ni < 4; ++ni)
      #pragma unroll
      for (int r = 0; r < 8; ++r) {
        int row = m0 + mi * 16 + half * 8 + r;
        int col = n0 + ni * 16 + lrow;
        float v = acc[mi][ni][r];
        if (epi == 1) v = v * fast_sigmoid(v);   // silu
        C[(size_t)row * N + col] = v;
      }
}

// ---------------------------------------------------------------------------
// alpha/beta = sigmoid(x @ Wa + ba), sigmoid(x @ Wb + bb)   [ROWS x 16]
// ---------------------------------------------------------------------------
__global__ __launch_bounds__(256) void alphabeta_kernel(const float* __restrict__ x,
                                                        const float* __restrict__ Wa,
                                                        const float* __restrict__ ba,
                                                        const float* __restrict__ Wb,
                                                        const float* __restrict__ bb,
                                                        float* __restrict__ alpha,
                                                        float* __restrict__ beta) {
  int idx = blockIdx.x * 256 + threadIdx.x;
  if (idx >= ROWS * N_HEADS) return;
  int row = idx >> 4, h = idx & 15;
  const float* xr = x + (size_t)row * D_MODEL;
  float sa = 0.f, sb = 0.f;
  for (int d = 0; d < D_MODEL; ++d) {
    float xv = xr[d];
    sa = fmaf(xv, Wa[d * N_HEADS + h], sa);
    sb = fmaf(xv, Wb[d * N_HEADS + h], sb);
  }
  alpha[idx] = fast_sigmoid(sa + ba[h]);
  beta[idx]  = fast_sigmoid(sb + bb[h]);
}

// ---------------------------------------------------------------------------
// Delta-rule scan. One workgroup (64 threads = 2 waves) per (b,h).
// Thread e owns state column state[0..63][e] in registers. Chunks of 64
// timesteps are staged in LDS (pitch 65 -> conflict-free); q/k L2-normalize
// fused into the chunk stage.
// ---------------------------------------------------------------------------
#define CH 64
#define PITCH 65
__global__ __launch_bounds__(64) void scan_kernel(const float* __restrict__ q,
                                                  const float* __restrict__ k,
                                                  const float* __restrict__ v,
                                                  const float* __restrict__ alpha,
                                                  const float* __restrict__ beta,
                                                  float* __restrict__ y) {
  int bh = blockIdx.x;
  int b = bh >> 4, h = bh & 15;
  int tid = threadIdx.x;

  __shared__ float qs[CH * PITCH];
  __shared__ float ks[CH * PITCH];
  __shared__ float vs[CH * PITCH];
  __shared__ float as_[CH];
  __shared__ float bs_[CH];

  float s[HEAD_DIM];
  #pragma unroll
  for (int d = 0; d < HEAD_DIM; ++d) s[d] = 0.f;

  for (int c0 = 0; c0 < SEQ; c0 += CH) {
    // stage chunk (coalesced: 64 lanes x contiguous head-dim)
    for (int t = 0; t < CH; ++t) {
      size_t base = ((size_t)(b * SEQ + c0 + t) * N_HEADS + h) * HEAD_DIM + tid;
      qs[t * PITCH + tid] = q[base];
      ks[t * PITCH + tid] = k[base];
      vs[t * PITCH + tid] = v[base];
    }
    {
      size_t ab = (size_t)(b * SEQ + c0 + tid) * N_HEADS + h;
      as_[tid] = alpha[ab];
      bs_[tid] = beta[ab];
    }
    __syncthreads();

    // fused L2 normalize of q,k: thread tid handles timestep tid
    {
      float ssq = 0.f, ssk = 0.f;
      #pragma unroll
      for (int d = 0; d < HEAD_DIM; ++d) {
        float qv = qs[tid * PITCH + d]; ssq = fmaf(qv, qv, ssq);
        float kv = ks[tid * PITCH + d]; ssk = fmaf(kv, kv, ssk);
      }
      float qn = __builtin_amdgcn_rcpf(fmaxf(__fsqrt_rn(ssq), 1e-12f));
      float kn = __builtin_amdgcn_rcpf(fmaxf(__fsqrt_rn(ssk), 1e-12f));
      #pragma unroll
      for (int d = 0; d < HEAD_DIM; ++d) {
        qs[tid * PITCH + d] *= qn;
        ks[tid * PITCH + d] *= kn;
      }
    }
    __syncthreads();

    for (int t = 0; t < CH; ++t) {
      float al = as_[t], be = bs_[t];
      float pred = 0.f;
      #pragma unroll
      for (int d = 0; d < HEAD_DIM; ++d)
        pred = fmaf(s[d], ks[t * PITCH + d], pred);       // LDS broadcast reads
      float bee = be * (vs[t * PITCH + tid] - pred);
      float out = 0.f;
      #pragma unroll
      for (int d = 0; d < HEAD_DIM; ++d) {
        float kv = ks[t * PITCH + d];
        s[d] = fmaf(al, s[d], bee * kv);
        out = fmaf(s[d], qs[t * PITCH + d], out);
      }
      y[((size_t)(b * SEQ + c0 + t) * N_HEADS + h) * HEAD_DIM + tid] = out;
    }
    __syncthreads();
  }
}

// ---------------------------------------------------------------------------
// z = rmsnorm(y, g_norm) * gate  -> bf16 (gate already has silu applied)
// One workgroup of 256 per row.
// ---------------------------------------------------------------------------
__global__ __launch_bounds__(256) void rmsnorm_gate_kernel(const float* __restrict__ y,
                                                           const float* __restrict__ gate,
                                                           const float* __restrict__ gnorm,
                                                           unsigned short* __restrict__ zb) {
  int row = blockIdx.x;
  __shared__ float red[256];
  const float* yr = y + (size_t)row * D_MODEL;
  float ss = 0.f;
  for (int d = threadIdx.x; d < D_MODEL; d += 256) {
    float vv = yr[d];
    ss = fmaf(vv, vv, ss);
  }
  red[threadIdx.x] = ss;
  __syncthreads();
  for (int off = 128; off > 0; off >>= 1) {
    if ((int)threadIdx.x < off) red[threadIdx.x] += red[threadIdx.x + off];
    __syncthreads();
  }
  float rms = __frsqrt_rn(red[0] / (float)D_MODEL + 1e-5f);
  for (int d = threadIdx.x; d < D_MODEL; d += 256) {
    float z = yr[d] * rms * gnorm[d] * gate[(size_t)row * D_MODEL + d];
    zb[(size_t)row * D_MODEL + d] = f2bf(z);
  }
}

// ---------------------------------------------------------------------------
extern "C" void kernel_launch(void* const* d_in, const int* in_sizes, int n_in,
                              void* d_out, int out_size, void* d_ws, size_t ws_size,
                              hipStream_t stream) {
  (void)in_sizes; (void)n_in; (void)out_size; (void)ws_size;

  const float* x  = (const float*)d_in[0];
  const float* Wq = (const float*)d_in[1];
  const float* Wk = (const float*)d_in[2];
  const float* Wv = (const float*)d_in[3];
  const float* Wa = (const float*)d_in[4];
  const float* ba = (const float*)d_in[5];
  const float* Wb = (const float*)d_in[6];
  const float* bb = (const float*)d_in[7];
  const float* Wg = (const float*)d_in[8];
  const float* Wo = (const float*)d_in[9];
  const float* gn = (const float*)d_in[10];
  float* out = (float*)d_out;

  // workspace layout (bytes)
  const size_t XB = (size_t)ROWS * D_MODEL * 2;      // 16 MB (bf16 activations)
  const size_t WT = (size_t)D_MODEL * D_MODEL * 2;   //  2 MB (bf16 transposed weights)
  const size_t FM = (size_t)ROWS * D_MODEL * 4;      // 32 MB (fp32 matrices)
  const size_t AB = (size_t)ROWS * N_HEADS * 4;      // 512 KB

  char* ws = (char*)d_ws;
  unsigned short* xb  = (unsigned short*)(ws);
  unsigned short* wqt = (unsigned short*)(ws + XB);
  unsigned short* wkt = (unsigned short*)(ws + XB + 1 * WT);
  unsigned short* wvt = (unsigned short*)(ws + XB + 2 * WT);
  unsigned short* wgt = (unsigned short*)(ws + XB + 3 * WT);
  unsigned short* wot = (unsigned short*)(ws + XB + 4 * WT);
  float* qf   = (float*)(ws + XB + 5 * WT);
  float* kf   = (float*)(ws + XB + 5 * WT + 1 * FM);
  float* vf   = (float*)(ws + XB + 5 * WT + 2 * FM);
  float* gf   = (float*)(ws + XB + 5 * WT + 3 * FM);
  float* yf   = (float*)(ws + XB + 5 * WT + 4 * FM);
  unsigned short* zb = (unsigned short*)(ws + XB + 5 * WT + 5 * FM);
  float* alf  = (float*)(ws + 2 * XB + 5 * WT + 5 * FM);
  float* bef  = (float*)(ws + 2 * XB + 5 * WT + 5 * FM + AB);

  // 1) x -> bf16
  {
    int n = ROWS * D_MODEL;
    cvt_f32_bf16_kernel<<<(n + 255) / 256, 256, 0, stream>>>(x, xb, n);
  }

  // 2) transpose + convert the five 1024x1024 weights
  {
    dim3 g(D_MODEL / 32, D_MODEL / 32);
    transpose_cvt_kernel<<<g, 256, 0, stream>>>(Wq, wqt, D_MODEL, D_MODEL);
    transpose_cvt_kernel<<<g, 256, 0, stream>>>(Wk, wkt, D_MODEL, D_MODEL);
    transpose_cvt_kernel<<<g, 256, 0, stream>>>(Wv, wvt, D_MODEL, D_MODEL);
    transpose_cvt_kernel<<<g, 256, 0, stream>>>(Wg, wgt, D_MODEL, D_MODEL);
    transpose_cvt_kernel<<<g, 256, 0, stream>>>(Wo, wot, D_MODEL, D_MODEL);
  }

  // 3) WMMA GEMMs: q, k, v, gate(silu)
  {
    dim3 g(ROWS / 128, D_MODEL / 128);
    gemm_bf16_kernel<<<g, 128, 0, stream>>>(xb, wqt, qf, ROWS, D_MODEL, D_MODEL, 0);
    gemm_bf16_kernel<<<g, 128, 0, stream>>>(xb, wkt, kf, ROWS, D_MODEL, D_MODEL, 0);
    gemm_bf16_kernel<<<g, 128, 0, stream>>>(xb, wvt, vf, ROWS, D_MODEL, D_MODEL, 0);
    gemm_bf16_kernel<<<g, 128, 0, stream>>>(xb, wgt, gf, ROWS, D_MODEL, D_MODEL, 1);
  }

  // 4) alpha / beta
  {
    int n = ROWS * N_HEADS;
    alphabeta_kernel<<<(n + 255) / 256, 256, 0, stream>>>(x, Wa, ba, Wb, bb, alf, bef);
  }

  // 5) sequential delta-rule scan (one WG per (b,h))
  scan_kernel<<<BATCH * N_HEADS, 64, 0, stream>>>(qf, kf, vf, alf, bef, yf);

  // 6) rmsnorm * gate -> bf16
  rmsnorm_gate_kernel<<<ROWS, 256, 0, stream>>>(yf, gf, gn, zb);

  // 7) final WMMA GEMM -> d_out
  {
    dim3 g(ROWS / 128, D_MODEL / 128);
    gemm_bf16_kernel<<<g, 128, 0, stream>>>(zb, wot, out, ROWS, D_MODEL, D_MODEL, 0);
  }
}
